// MoECrossAttention_27685359190686
// MI455X (gfx1250) — compile-verified, matrix operand order checked by reading, not run on hardware
//
#include <hip/hip_runtime.h>

// ---------------------------------------------------------------------------
// Fixed problem shapes
// ---------------------------------------------------------------------------
#define BATCH 4
#define NSEQ  2048
#define MSEQ  2048
#define CDIM  256
#define NHEAD 8
#define DHEAD 32
#define HDIM  1024
#define MROWS (BATCH * NSEQ)   // 8192

typedef unsigned int u32;
typedef unsigned short u16;
typedef unsigned long long u64;
typedef __attribute__((ext_vector_type(16))) __bf16 v16bf;
typedef __attribute__((ext_vector_type(8)))  float  v8f;
typedef __attribute__((ext_vector_type(8)))  u32    v8u;
typedef __attribute__((ext_vector_type(4)))  u32    v4u;
typedef __attribute__((ext_vector_type(4)))  u32    u32x4;
typedef __attribute__((ext_vector_type(8)))  int    i32x8;
typedef __attribute__((ext_vector_type(4)))  int    i32x4;

// float -> bf16 (round to nearest even)
static __device__ __forceinline__ u16 f2b(float f) {
    union { float f; u32 u; } x; x.f = f;
    u32 r = x.u + 0x7FFFu + ((x.u >> 16) & 1u);
    return (u16)(r >> 16);
}

static __device__ __forceinline__ v8f wmma_bf16(v16bf a, v16bf b, v8f c) {
    return __builtin_amdgcn_wmma_f32_16x16x32_bf16(
        false, a, false, b, (short)0, c, false, false);
}

// Load one 16x32 bf16 fragment in the CDNA5 16-bit A-matrix layout.
// base points at (row 0, k 0) of the tile, SU = row stride in dwords.
template <int SU>
static __device__ __forceinline__ v16bf load_frag(const u32* base, int lane) {
    const u32* p = base + (lane & 15) * SU + ((lane >> 4) << 2);
    v4u lo = *(const v4u*)(p);
    v4u hi = *(const v4u*)(p + 8);
    v8u u = { lo.x, lo.y, lo.z, lo.w, hi.x, hi.y, hi.z, hi.w };
    return __builtin_bit_cast(v16bf, u);
}

// ---------------------------------------------------------------------------
// Tensor Data Mover: 2D bf16 tile (rows x 32 elems, 64B rows) global -> LDS,
// LDS row stride 80B via pad_enable (pad 4 dwords after every 16 dwords).
// Descriptor packing per CDNA5 ISA sec 8.3-8.6. Issued by one wave, EXEC
// ignored; completion tracked with TENSORcnt (in-order per wave).
// 6-arg builtin: (uint32x4 g0, int32x8 g1, int32x4, int32x4, int32x8, i32 cpol)
// ---------------------------------------------------------------------------
static __device__ __forceinline__ void tdm_load_2d(
    u32 lds_byte_addr, const void* gsrc, u32 elem_stride, u32 rows) {
    u64 ga = (u64)(uintptr_t)gsrc;
    u32x4 g0;
    g0.x = 1u;                                    // count=1, no gather
    g0.y = lds_byte_addr;                         // lds_addr
    g0.z = (u32)ga;                               // global_addr[31:0]
    g0.w = (u32)(ga >> 32) | (2u << 30);          // global_addr[56:32] | type=2
    const u32 TD = 0x100000u;                     // generous tensor dims (no OOB)
    u32 gi0 = (1u << 16)                          // data_size = 2 bytes
            | (1u << 20)                          // pad_enable
            | (3u << 22)                          // pad_interval: 16 dwords (64B)
            | (3u << 25);                         // pad_amount: 4 dwords (16B)
    u32 gi1 = (TD & 0xFFFFu) << 16;               // tensor_dim0 lo
    u32 gi2 = (TD >> 16) | ((TD & 0xFFFFu) << 16);// tensor_dim0 hi | tensor_dim1 lo
    u32 gi3 = (TD >> 16) | (32u << 16);           // tensor_dim1 hi | tile_dim0 = 32
    u32 gi4 = rows;                               // tile_dim1 = rows, tile_dim2 = 0
    u32 gi5 = elem_stride;                        // tensor_dim0_stride[31:0]
    i32x8 g1 = { (int)gi0, (int)gi1, (int)gi2, (int)gi3,
                 (int)gi4, (int)gi5, 0, 0 };
    i32x4 gz = { 0, 0, 0, 0 };
    i32x8 gz8 = { 0, 0, 0, 0, 0, 0, 0, 0 };
    __builtin_amdgcn_tensor_load_to_lds(g0, g1, gz, gz, gz8, 0);
}

// ---------------------------------------------------------------------------
// Conversion kernels
// ---------------------------------------------------------------------------
__global__ __launch_bounds__(256) void cvt_f32_bf16(const float* __restrict__ in,
                                                    u16* __restrict__ out, int n) {
    int i = blockIdx.x * 256 + threadIdx.x;
    if (i < n) out[i] = f2b(in[i]);
}

// in: [K][N] f32  ->  out: [N][K] bf16 (transposed)
__global__ __launch_bounds__(256) void cvt_tr_bf16(const float* __restrict__ in,
                                                   u16* __restrict__ out, int K, int N) {
    int i = blockIdx.x * 256 + threadIdx.x;
    if (i < K * N) {
        int n = i / K, k = i % K;
        out[i] = f2b(in[(size_t)k * N + n]);
    }
}

// ---------------------------------------------------------------------------
// bf16 WMMA GEMM: D[8192 x Ncols] = A[8192 x K] * B[K x Ncols], BT = [Ncols][K].
// 256 threads = 8 waves; block tile 128x128; wave tile 32x64 (8 WMMA / K-step).
// TDM double-buffered LDS staging (wave 0 drives the DMA pipeline).
// Epilogue modes: 0 = Q scatter, 1 = KV scatter (V transposed), 2 = GELU->bf16,
//                 3 = bias + token-type-masked f32 store.
// ---------------------------------------------------------------------------
__global__ __launch_bounds__(256) void gemm_bf16(
    const u16* __restrict__ A, const u16* __restrict__ BT,
    const float* __restrict__ bias, int Ncols, int K, int mode,
    const int* __restrict__ tt, int expert,
    u16* __restrict__ outB0, u16* __restrict__ outB1, float* __restrict__ outF) {

    __shared__ __align__(16) u16 As[2][128 * 40];
    __shared__ __align__(16) u16 Bs[2][128 * 40];

    const int bx = blockIdx.x;            // col block (128 cols)
    const int by = blockIdx.y;            // row block (128 rows)
    const int t = threadIdx.x;
    const int w = t >> 5, lane = t & 31;
    const int wr = w >> 1, wc = w & 1;    // 4x2 waves -> 32x64 tiles

    const int Ksteps = K >> 5;

    v8f zero = { 0.f, 0.f, 0.f, 0.f, 0.f, 0.f, 0.f, 0.f };
    v8f acc[2][4];
    for (int i = 0; i < 2; ++i)
        for (int j = 0; j < 4; ++j) acc[i][j] = zero;

    if (w == 0) {   // prologue: stage 0 into buffer 0
        tdm_load_2d((u32)(uintptr_t)(void*)&As[0][0],
                    &A [(size_t)(by * 128) * K], (u32)K, 128);
        tdm_load_2d((u32)(uintptr_t)(void*)&Bs[0][0],
                    &BT[(size_t)(bx * 128) * K], (u32)K, 128);
    }

    for (int s = 0; s < Ksteps; ++s) {
        __syncthreads();                  // next buffer free (prev compute done)
        if (w == 0) {
            if (s + 1 < Ksteps) {
                int buf = (s + 1) & 1;
                tdm_load_2d((u32)(uintptr_t)(void*)&As[buf][0],
                            &A [(size_t)(by * 128) * K + (s + 1) * 32], (u32)K, 128);
                tdm_load_2d((u32)(uintptr_t)(void*)&Bs[buf][0],
                            &BT[(size_t)(bx * 128) * K + (s + 1) * 32], (u32)K, 128);
                __builtin_amdgcn_s_wait_tensorcnt(2);   // stage s resident
            } else {
                __builtin_amdgcn_s_wait_tensorcnt(0);
            }
        }
        __syncthreads();                  // buffer s&1 visible to all waves

        const u32* ap = (const u32*)&As[s & 1][0];
        const u32* bp = (const u32*)&Bs[s & 1][0];
        v16bf a0 = load_frag<20>(ap + (wr * 32 +  0) * 20, lane);
        v16bf a1 = load_frag<20>(ap + (wr * 32 + 16) * 20, lane);
        for (int j = 0; j < 4; ++j) {
            v16bf bj = load_frag<20>(bp + (wc * 64 + j * 16) * 20, lane);
            acc[0][j] = wmma_bf16(a0, bj, acc[0][j]);
            acc[1][j] = wmma_bf16(a1, bj, acc[1][j]);
        }
    }

    // Epilogue.  C/D layout: lane (mod 16) = column, VGPR v = row (lanes>=16 -> +8).
    for (int mi = 0; mi < 2; ++mi)
    for (int mj = 0; mj < 4; ++mj)
    for (int v = 0; v < 8; ++v) {
        int row = by * 128 + wr * 32 + mi * 16 + v + ((lane >> 4) << 3);
        int col = bx * 128 + wc * 64 + mj * 16 + (lane & 15);
        float val = acc[mi][mj][v];
        if (mode == 0) {
            int b = row >> 11, n = row & 2047;
            int h = col >> 5, dd = col & 31;
            outB0[(((size_t)(b * NHEAD + h) * NSEQ + n) << 5) + dd] = f2b(val);
        } else if (mode == 1) {
            int b = row >> 11, m = row & 2047;
            if (col < CDIM) {
                int h = col >> 5, dd = col & 31;
                outB0[(((size_t)(b * NHEAD + h) * MSEQ + m) << 5) + dd] = f2b(val);
            } else {
                int c2 = col - CDIM;
                int h = c2 >> 5, dd = c2 & 31;
                outB1[((size_t)(b * NHEAD + h) * DHEAD + dd) * MSEQ + m] = f2b(val);
            }
        } else if (mode == 2) {
            float xg = val + bias[col];
            float g = 0.5f * xg * (1.0f + erff(xg * 0.70710678118654752f));
            outB0[(size_t)row * Ncols + col] = f2b(g);
        } else { // mode 3
            if (tt[row] == expert)
                outF[(size_t)row * Ncols + col] = val + bias[col];
        }
    }
}

// ---------------------------------------------------------------------------
// Flash attention: per (b,h) head, 64 query rows per block (4 waves x 16 rows).
// q: [B,H,N,32] bf16, k: [B,H,M,32] bf16, vT: [B,H,32,M] bf16,
// ctx out: [B,N,C] bf16 (column offset h*32).
// K/V chunks staged via TDM, double-buffered (wave 0 drives the pipeline).
// ---------------------------------------------------------------------------
__global__ __launch_bounds__(128) void attn_kernel(
    const u16* __restrict__ qb, const u16* __restrict__ kb,
    const u16* __restrict__ vbT, u16* __restrict__ ctx) {

    __shared__ __align__(16) u16 Ks[2][32 * 40];
    __shared__ __align__(16) u16 Vs[2][32 * 40];
    __shared__ __align__(16) u16 Ps[4 * 16 * 40];

    const int bh = blockIdx.x;                 // b*8 + h
    const int n0 = blockIdx.y * 64;
    const int t = threadIdx.x;
    const int w = t >> 5, lane = t & 31;
    const float scale = 0.17677669529663687f;  // 1/sqrt(32)
    const int Msteps = MSEQ / 32;              // 64

    const u32* qp = (const u32*)qb;
    v16bf qa = load_frag<16>(qp + (size_t)(bh * NSEQ + n0 + w * 16) * 16, lane);

    v8f zero = { 0.f, 0.f, 0.f, 0.f, 0.f, 0.f, 0.f, 0.f };
    v8f o0 = zero, o1 = zero;
    float rmax[8], rsum[8];
    for (int v = 0; v < 8; ++v) { rmax[v] = -1e30f; rsum[v] = 0.f; }

    const u16* kbase = &kb [(size_t)(bh * MSEQ) * 32];
    const u16* vbase = &vbT[(size_t)(bh * DHEAD) * MSEQ];

    if (w == 0) {   // prologue: stage chunk 0 into buffer 0
        tdm_load_2d((u32)(uintptr_t)(void*)&Ks[0][0], kbase, 32u, 32);
        tdm_load_2d((u32)(uintptr_t)(void*)&Vs[0][0], vbase, (u32)MSEQ, 32);
    }

    for (int s = 0; s < Msteps; ++s) {
        __syncthreads();                       // next buffer free (prev compute done)
        if (w == 0) {
            if (s + 1 < Msteps) {
                int buf = (s + 1) & 1;
                tdm_load_2d((u32)(uintptr_t)(void*)&Ks[buf][0],
                            kbase + (size_t)(s + 1) * 32 * 32, 32u, 32);
                tdm_load_2d((u32)(uintptr_t)(void*)&Vs[buf][0],
                            vbase + (s + 1) * 32, (u32)MSEQ, 32);
                __builtin_amdgcn_s_wait_tensorcnt(2);   // chunk s resident
            } else {
                __builtin_amdgcn_s_wait_tensorcnt(0);
            }
        }
        __syncthreads();                       // tiles visible

        const u32* kp = (const u32*)&Ks[s & 1][0];
        const u32* vp = (const u32*)&Vs[s & 1][0];
        v16bf bk0 = load_frag<20>(kp,           lane);
        v16bf bk1 = load_frag<20>(kp + 16 * 20, lane);
        v8f s0 = wmma_bf16(qa, bk0, zero);
        v8f s1 = wmma_bf16(qa, bk1, zero);

        for (int v = 0; v < 8; ++v) {
            float a = s0[v] * scale, b = s1[v] * scale;
            float tmax = fmaxf(a, b);
            for (int m = 8; m >= 1; m >>= 1) tmax = fmaxf(tmax, __shfl_xor(tmax, m, 32));
            float nm = fmaxf(rmax[v], tmax);
            float p0 = __expf(a - nm), p1 = __expf(b - nm);
            float corr = __expf(rmax[v] - nm);
            rmax[v] = nm;
            float ps = p0 + p1;
            for (int m = 8; m >= 1; m >>= 1) ps += __shfl_xor(ps, m, 32);
            rsum[v] = rsum[v] * corr + ps;
            o0[v] *= corr;
            o1[v] *= corr;
            int prow = w * 16 + v + ((lane >> 4) << 3);     // C-layout row
            Ps[prow * 40 + (lane & 15)]      = f2b(p0);
            Ps[prow * 40 + 16 + (lane & 15)] = f2b(p1);
        }
        asm volatile("s_wait_dscnt 0" ::: "memory");        // same-wave P order

        v16bf pa  = load_frag<20>((const u32*)Ps + w * 16 * 20, lane);
        v16bf bv0 = load_frag<20>(vp,           lane);
        v16bf bv1 = load_frag<20>(vp + 16 * 20, lane);
        o0 = wmma_bf16(pa, bv0, o0);
        o1 = wmma_bf16(pa, bv1, o1);
    }

    const int b = bh >> 3, h = bh & 7;
    for (int v = 0; v < 8; ++v) {
        int n = n0 + w * 16 + v + ((lane >> 4) << 3);
        float inv = 1.0f / rsum[v];
        size_t base = ((size_t)(b * NSEQ + n)) * CDIM + h * DHEAD;
        int dd = lane & 15;
        ctx[base + dd]      = f2b(o0[v] * inv);
        ctx[base + 16 + dd] = f2b(o1[v] * inv);
    }
}

// ---------------------------------------------------------------------------
// Host-side orchestration
// ---------------------------------------------------------------------------
extern "C" void kernel_launch(void* const* d_in, const int* in_sizes, int n_in,
                              void* d_out, int out_size, void* d_ws, size_t ws_size,
                              hipStream_t stream) {
    (void)in_sizes; (void)n_in; (void)out_size; (void)ws_size;

    const float* x   = (const float*)d_in[0];
    const float* y   = (const float*)d_in[1];
    const int*   tt  = (const int*)d_in[2];
    const float* Wq  = (const float*)d_in[3];
    const float* Wkv = (const float*)d_in[4];
    const float* Ws1 = (const float*)d_in[5];
    const float* bs1 = (const float*)d_in[6];
    const float* Ws2 = (const float*)d_in[7];
    const float* bs2 = (const float*)d_in[8];
    const float* Wl1 = (const float*)d_in[9];
    const float* bl1 = (const float*)d_in[10];
    const float* Wl2 = (const float*)d_in[11];
    const float* bl2 = (const float*)d_in[12];
    float* out = (float*)d_out;

    char* ws = (char*)d_ws;
    const size_t MB = 1u << 20;
    u16* qb   = (u16*)(ws + 0 * MB);        //  4 MB  [B,H,N,32]
    u16* kb   = (u16*)(ws + 4 * MB);        //  4 MB  [B,H,M,32]
    u16* vbT  = (u16*)(ws + 8 * MB);        //  4 MB  [B,H,32,M]
    u16* ctx  = (u16*)(ws + 12 * MB);       //  4 MB  [B*N, C]
    u16* hid  = (u16*)(ws + 16 * MB);       // 16 MB  [B*N, Hd]
    u16* xb   = (u16*)(ws + 32 * MB);       //  4 MB
    u16* yb   = (u16*)(ws + 36 * MB);       //  4 MB
    u16* WqT  = (u16*)(ws + 40 * MB);                       // [256][256]
    u16* WkvT = (u16*)((char*)WqT  + 256 * 256 * 2);        // [512][256]
    u16* Ws1T = (u16*)((char*)WkvT + 512 * 256 * 2);        // [1024][256]
    u16* Ws2T = (u16*)((char*)Ws1T + 1024 * 256 * 2);       // [256][1024]
    u16* Wl1T = (u16*)((char*)Ws2T + 256 * 1024 * 2);       // [1024][256]
    u16* Wl2T = (u16*)((char*)Wl1T + 1024 * 256 * 2);       // [256][1024]

    // 1) precision conversion (+ weight transposes)
    cvt_f32_bf16<<<MROWS * CDIM / 256, 256, 0, stream>>>(x, xb, MROWS * CDIM);
    cvt_f32_bf16<<<MROWS * CDIM / 256, 256, 0, stream>>>(y, yb, MROWS * CDIM);
    cvt_tr_bf16<<<(256 * 256) / 256, 256, 0, stream>>>(Wq,  WqT,  256, 256);
    cvt_tr_bf16<<<(256 * 512) / 256, 256, 0, stream>>>(Wkv, WkvT, 256, 512);
    cvt_tr_bf16<<<(256 * 1024) / 256, 256, 0, stream>>>(Ws1, Ws1T, 256, 1024);
    cvt_tr_bf16<<<(1024 * 256) / 256, 256, 0, stream>>>(Ws2, Ws2T, 1024, 256);
    cvt_tr_bf16<<<(256 * 1024) / 256, 256, 0, stream>>>(Wl1, Wl1T, 256, 1024);
    cvt_tr_bf16<<<(1024 * 256) / 256, 256, 0, stream>>>(Wl2, Wl2T, 1024, 256);

    // 2) projections
    gemm_bf16<<<dim3(CDIM / 128, MROWS / 128), 256, 0, stream>>>(
        xb, WqT, nullptr, CDIM, CDIM, 0, nullptr, 0, qb, nullptr, nullptr);
    gemm_bf16<<<dim3(2 * CDIM / 128, MROWS / 128), 256, 0, stream>>>(
        yb, WkvT, nullptr, 2 * CDIM, CDIM, 1, nullptr, 0, kb, vbT, nullptr);

    // 3) flash attention (grid: 32 heads x 32 query blocks)
    attn_kernel<<<dim3(BATCH * NHEAD, NSEQ / 64), 128, 0, stream>>>(qb, kb, vbT, ctx);

    // 4) MoE MLP: expert S then expert L, experts write disjoint rows of d_out
    gemm_bf16<<<dim3(HDIM / 128, MROWS / 128), 256, 0, stream>>>(
        ctx, Ws1T, bs1, HDIM, CDIM, 2, nullptr, 0, hid, nullptr, nullptr);
    gemm_bf16<<<dim3(CDIM / 128, MROWS / 128), 256, 0, stream>>>(
        hid, Ws2T, bs2, CDIM, HDIM, 3, tt, 0, nullptr, nullptr, out);
    gemm_bf16<<<dim3(HDIM / 128, MROWS / 128), 256, 0, stream>>>(
        ctx, Wl1T, bl1, HDIM, CDIM, 2, nullptr, 0, hid, nullptr, nullptr);
    gemm_bf16<<<dim3(CDIM / 128, MROWS / 128), 256, 0, stream>>>(
        hid, Wl2T, bl2, CDIM, HDIM, 3, tt, 1, nullptr, nullptr, out);
}
